// GuidedAnchor_69071664054290
// MI455X (gfx1250) — matrix operand order repaired and checked by reading, not compiled, not run on hardware
//
#include <hip/hip_runtime.h>
#include <hip/hip_bf16.h>

typedef __attribute__((ext_vector_type(16))) __bf16 v16bf;
typedef __attribute__((ext_vector_type(8)))  __bf16 v8bf;
typedef __attribute__((ext_vector_type(8)))  float  v8f;

#define AST 40  // A-tile LDS row stride in __bf16 elements (80 bytes, 16B aligned, conflict-free)

// ---------------------------------------------------------------------------
// Pack adapt_w [256][256][3][3] f32 -> bf16 WMMA-B layout.
// K index: k = kk*256 + c  (kk = ky*3+kx). Tiles: kb (72, K=32 each) x nb (16, N=16 each).
// Within a tile: lane l holds column N = l%16, K_local = (l>>4)*16 + e, e = 0..15.
// Flat: Bp[ ((kb*16+nb)*32 + lane)*16 + e ]
// ---------------------------------------------------------------------------
__global__ void ga_pack_w(const float* __restrict__ aw, __bf16* __restrict__ bp) {
  int t = blockIdx.x * blockDim.x + threadIdx.x;
  if (t >= 72 * 16 * 32) return;
  int lane = t & 31;
  int nb   = (t >> 5) & 15;
  int kb   = t >> 9;
  int kk    = kb >> 3;
  int cbase = (kb & 7) * 32;
  int o     = nb * 16 + (lane & 15);
  int khalf = lane >> 4;
  __bf16* dst = bp + (size_t)t * 16;
#pragma unroll
  for (int e = 0; e < 16; ++e) {
    int c = cbase + khalf * 16 + e;
    dst[e] = (__bf16)aw[(size_t)o * 2304 + c * 9 + kk];
  }
}

// ---------------------------------------------------------------------------
// Head: loc = sigmoid(conv3x3 -> 1ch), sh = conv3x3 -> 2ch,
// reformed = sigma*stride*exp(sh), off = 1x1 conv of sh -> 72 ch (to workspace).
// One thread per output pixel.
// ---------------------------------------------------------------------------
__global__ void ga_head(const float* __restrict__ f,
                        const float* __restrict__ lw, const float* __restrict__ lb,
                        const float* __restrict__ sw, const float* __restrict__ sb,
                        const float* __restrict__ ow,
                        float* __restrict__ out_loc, float* __restrict__ out_sh,
                        float* __restrict__ out_ref, float* __restrict__ off_ws,
                        int H, int W, float sigma_stride) {
  int HW = H * W;
  int P  = 2 * HW;
  int p  = blockIdx.x * blockDim.x + threadIdx.x;
  if (p >= P) return;
  int n = p / HW, rem = p % HW, h = rem / W, w = rem % W;

  float accL = 0.f, acc0 = 0.f, acc1 = 0.f;
  for (int c = 0; c < 256; ++c) {
    const float* fc  = f + ((size_t)(n * 256 + c)) * HW;
    const float* lwc = lw + c * 9;
    const float* s0c = sw + c * 9;
    const float* s1c = sw + 2304 + c * 9;
#pragma unroll
    for (int ky = 0; ky < 3; ++ky) {
      int y = h - 1 + ky;
      if ((unsigned)y >= (unsigned)H) continue;
#pragma unroll
      for (int kx = 0; kx < 3; ++kx) {
        int x = w - 1 + kx;
        if ((unsigned)x >= (unsigned)W) continue;
        float v = fc[y * W + x];
        int t = ky * 3 + kx;
        accL = fmaf(v, lwc[t], accL);
        acc0 = fmaf(v, s0c[t], acc0);
        acc1 = fmaf(v, s1c[t], acc1);
      }
    }
  }
  accL += lb[0]; acc0 += sb[0]; acc1 += sb[1];

  out_loc[(size_t)n * HW + rem]            = 1.f / (1.f + expf(-accL));
  out_sh[((size_t)(n * 2 + 0)) * HW + rem] = acc0;
  out_sh[((size_t)(n * 2 + 1)) * HW + rem] = acc1;
  out_ref[((size_t)(n * 2 + 0)) * HW + rem] = sigma_stride * expf(acc0);
  out_ref[((size_t)(n * 2 + 1)) * HW + rem] = sigma_stride * expf(acc1);

#pragma unroll 4
  for (int ch = 0; ch < 72; ++ch) {
    float v = ow[ch * 2 + 0] * acc0 + ow[ch * 2 + 1] * acc1;
    off_ws[((size_t)(n * 72 + ch)) * HW + rem] = v;
  }
}

// ---------------------------------------------------------------------------
// Deform conv implicit GEMM: M=32 pixels/block, N=256 channels/block, K=2304.
// K-step kb (32 wide) -> single (dg, kk): one bilinear plan per pixel.
// 8 waves: wave (wm,wn) = (wid&1, wid>>1): 16 rows x 64 cols -> 4 WMMAs/step.
// Double-buffered A tile in LDS -> one barrier per K-step; gather for step
// kb+1 overlaps the WMMAs of step kb. B fragments preloaded as a batch.
// ---------------------------------------------------------------------------
__global__ void __launch_bounds__(256) ga_dconv(const float* __restrict__ f,
                                                const float* __restrict__ off,
                                                const __bf16* __restrict__ Bp,
                                                float* __restrict__ out,
                                                int H, int W) {
  __shared__ __bf16 Als[2][32 * AST];
  int HW = H * W;
  int P  = 2 * HW;
  int tid  = threadIdx.x;
  int lane = tid & 31;
  int wid  = tid >> 5;
  int wm   = wid & 1;
  int wn   = wid >> 1;
  int r    = lane & 15;
  int hl   = lane >> 4;
  int mbase = blockIdx.x * 32;

  // gather role: 8 threads per pixel, 4 channels each
  int gm   = tid >> 3;
  int quad = tid & 7;
  int p = mbase + gm;
  int n = 0, h = 0, w = 0;
  size_t pix = 0;
  bool valid = (p < P);
  if (valid) { n = p / HW; int rem = p % HW; h = rem / W; w = rem % W; pix = rem; }
  const float* fbatch = f + (size_t)n * 256 * HW;          // feature base for this pixel's image
  const float* obatch = off + (size_t)n * 72 * HW + pix;   // offset base for this pixel

  // bilinear im2col gather for K-step kb -> 4 channel values for this thread
  auto gather = [&](int kb, float* vals) {
    vals[0] = vals[1] = vals[2] = vals[3] = 0.f;
    if (!valid) return;
    int kk = kb >> 3;
    int cb = (kb & 7) * 32;
    int dg = cb >> 6;
    int och = (dg * 9 + kk) * 2;
    float dy = obatch[(size_t)och * HW];
    float dx = obatch[(size_t)(och + 1) * HW];
    float py = (float)(h - 1 + kk / 3) + dy;
    float px = (float)(w - 1 + kk % 3) + dx;
    float y0f = floorf(py), x0f = floorf(px);
    float ly = py - y0f, lx = px - x0f;
    int y0 = (int)y0f, x0 = (int)x0f;
    float w00 = (1.f - ly) * (1.f - lx), w01 = (1.f - ly) * lx;
    float w10 = ly * (1.f - lx),         w11 = ly * lx;
    bool vy0 = (unsigned)y0       < (unsigned)H;
    bool vy1 = (unsigned)(y0 + 1) < (unsigned)H;
    bool vx0 = (unsigned)x0       < (unsigned)W;
    bool vx1 = (unsigned)(x0 + 1) < (unsigned)W;
    const float* fb = fbatch + (size_t)(cb + quad * 4) * HW;
#pragma unroll
    for (int j = 0; j < 4; ++j) {
      const float* fc = fb + (size_t)j * HW;
      float v = 0.f;
      if (vy0 && vx0) v = fmaf(w00, fc[y0 * W + x0],           v);
      if (vy0 && vx1) v = fmaf(w01, fc[y0 * W + x0 + 1],       v);
      if (vy1 && vx0) v = fmaf(w10, fc[(y0 + 1) * W + x0],     v);
      if (vy1 && vx1) v = fmaf(w11, fc[(y0 + 1) * W + x0 + 1], v);
      vals[j] = v;
    }
  };

  v8f acc[4] = {};

  // prologue: stage K-step 0 into buffer 0
  {
    float vals[4];
    gather(0, vals);
    __bf16* dst = &Als[0][gm * AST + quad * 4];
    dst[0] = (__bf16)vals[0];
    dst[1] = (__bf16)vals[1];
    dst[2] = (__bf16)vals[2];
    dst[3] = (__bf16)vals[3];
  }
  __syncthreads();

  for (int kb = 0; kb < 72; ++kb) {
    int cur = kb & 1;
    const __bf16* arow = &Als[cur][(wm * 16 + r) * AST];

    // ---- batch-issue all 4 B fragment loads first (one wait covers all) ----
    const __bf16* bbase = Bp + (((size_t)(kb * 16 + wn * 4)) * 32 + lane) * 16;
    __builtin_prefetch(bbase + (size_t)16 * 32 * 16, 0, 3);
    v16bf bfr[4];
#pragma unroll
    for (int j = 0; j < 4; ++j) {
      const __bf16* bt = bbase + (size_t)j * 32 * 16;
      v8bf blo = *(const v8bf*)bt;
      v8bf bhi = *(const v8bf*)(bt + 8);
      bfr[j] = __builtin_shufflevector(blo, bhi, 0, 1, 2, 3, 4, 5, 6, 7,
                                                 8, 9, 10, 11, 12, 13, 14, 15);
    }

    // ---- issue next K-step's gather loads (overlap with WMMAs below) ----
    float vals[4] = {0.f, 0.f, 0.f, 0.f};
    if (kb + 1 < 72) gather(kb + 1, vals);

    // ---- A fragment from LDS (two b128 loads per lane) ----
    v8bf alo = *(const v8bf*)(arow + hl * 16);
    v8bf ahi = *(const v8bf*)(arow + hl * 16 + 8);
    v16bf a = __builtin_shufflevector(alo, ahi, 0, 1, 2, 3, 4, 5, 6, 7,
                                                8, 9, 10, 11, 12, 13, 14, 15);

#pragma unroll
    for (int j = 0; j < 4; ++j) {
      acc[j] = __builtin_amdgcn_wmma_f32_16x16x32_bf16(
          false, a, false, bfr[j], (short)0, acc[j], false, false);
    }

    // ---- stage next K-step into the other buffer; single barrier ----
    {
      __bf16* dst = &Als[cur ^ 1][gm * AST + quad * 4];
      dst[0] = (__bf16)vals[0];
      dst[1] = (__bf16)vals[1];
      dst[2] = (__bf16)vals[2];
      dst[3] = (__bf16)vals[3];
    }
    __syncthreads();
  }

  // ---- epilogue: ReLU + scatter to NCHW ----
#pragma unroll
  for (int j = 0; j < 4; ++j) {
    int o = wn * 64 + j * 16 + r;
#pragma unroll
    for (int e = 0; e < 8; ++e) {
      int ml = wm * 16 + hl * 8 + e;
      int pp = mbase + ml;
      if (pp < P) {
        int nn = pp / HW, rem = pp % HW;
        float v = acc[j][e];
        out[((size_t)(nn * 256 + o)) * HW + rem] = v > 0.f ? v : 0.f;
      }
    }
  }
}

// ---------------------------------------------------------------------------
extern "C" void kernel_launch(void* const* d_in, const int* in_sizes, int n_in,
                              void* d_out, int out_size, void* d_ws, size_t ws_size,
                              hipStream_t stream) {
  (void)in_sizes; (void)n_in; (void)out_size; (void)ws_size;
  const float* feat[4] = {(const float*)d_in[0], (const float*)d_in[1],
                          (const float*)d_in[2], (const float*)d_in[3]};
  const float* loc_w    = (const float*)d_in[4];
  const float* loc_b    = (const float*)d_in[5];
  const float* shape_w  = (const float*)d_in[6];
  const float* shape_b  = (const float*)d_in[7];
  const float* offset_w = (const float*)d_in[8];
  const float* adapt_w  = (const float*)d_in[9];
  float* out = (float*)d_out;

  const int   Hs[4] = {100, 50, 25, 13};
  const int   Ws[4] = {152, 76, 38, 19};
  const float St[4] = {8.f, 16.f, 32.f, 64.f};
  size_t HW[4], P[4];
  for (int l = 0; l < 4; ++l) { HW[l] = (size_t)Hs[l] * Ws[l]; P[l] = 2 * HW[l]; }

  // output offsets: loc0..3, sh0..3, ref0..3, adapt0..3
  size_t o_loc[4], o_sh[4], o_ref[4], o_ad[4], cur = 0;
  for (int l = 0; l < 4; ++l) { o_loc[l] = cur; cur += 2 * HW[l]; }
  for (int l = 0; l < 4; ++l) { o_sh[l]  = cur; cur += 4 * HW[l]; }
  for (int l = 0; l < 4; ++l) { o_ref[l] = cur; cur += 4 * HW[l]; }
  for (int l = 0; l < 4; ++l) { o_ad[l]  = cur; cur += 512 * HW[l]; }

  // workspace: packed B (bf16) then per-level offset maps (f32)
  __bf16* Bp    = (__bf16*)d_ws;
  float* offbuf = (float*)((char*)d_ws + (size_t)72 * 16 * 32 * 16 * sizeof(__bf16));
  size_t o_off[4]; cur = 0;
  for (int l = 0; l < 4; ++l) { o_off[l] = cur; cur += P[l] * 72; }

  ga_pack_w<<<(72 * 16 * 32 + 255) / 256, 256, 0, stream>>>(adapt_w, Bp);

  for (int l = 0; l < 4; ++l) {
    int blocks = (int)((P[l] + 255) / 256);
    ga_head<<<blocks, 256, 0, stream>>>(feat[l], loc_w, loc_b, shape_w, shape_b,
                                        offset_w, out + o_loc[l], out + o_sh[l],
                                        out + o_ref[l], offbuf + o_off[l],
                                        Hs[l], Ws[l], 8.f * St[l]);
  }
  for (int l = 0; l < 4; ++l) {
    int blocks = (int)((P[l] + 31) / 32);
    ga_dconv<<<blocks, 256, 0, stream>>>(feat[l], offbuf + o_off[l], Bp,
                                         out + o_ad[l], Hs[l], Ws[l]);
  }
}
